// Supernet_69398081569083
// MI455X (gfx1250) — matrix operand order repaired and checked by reading, not compiled, not run on hardware
//
#include <hip/hip_runtime.h>
#include <hip/hip_bf16.h>

typedef __attribute__((ext_vector_type(16))) _Float16 v16h;
typedef __attribute__((ext_vector_type(4)))  _Float16 v4h;
typedef __attribute__((ext_vector_type(8)))  float    v8f;

#define TPB 256
static inline int cdiv(int a, int b) { return (a + b - 1) / b; }

// =====================================================================
// pack_x: f32 NCHW activations -> packed f16 K-major matrix Xp[M][Kp]
//   Xp[m][kk] = act(X[b, kk, ho*stride+off, wo*stride+off]),  0 if kk>=Cin
// Each thread writes 4 consecutive kk (one 8-byte store).
// =====================================================================
__global__ __launch_bounds__(256) void pack_x_kernel(
    const float* __restrict__ X, _Float16* __restrict__ Xp,
    int Bn, int Cin, int Hin, int Win, int Hout, int Wout,
    int stride, int off, int relu, int Kp)
{
  const int Kq = Kp >> 2;
  int idx = blockIdx.x * blockDim.x + threadIdx.x;
  int total = Bn * Hout * Wout * Kq;
  if (idx >= total) return;
  int kq = idx % Kq;
  int m  = idx / Kq;
  int kk0 = kq << 2;
  const int HWo = Hout * Wout;
  int b  = m / HWo;
  int sp = m - b * HWo;
  int ho = sp / Wout, wo = sp - (sp / Wout) * Wout;
  int hi = ho * stride + off, wi = wo * stride + off;
  const float* xp = X + ((size_t)b * Cin * Hin + hi) * Win + wi;
  const size_t chs = (size_t)Hin * Win;

  v4h o;
#pragma unroll
  for (int i = 0; i < 4; ++i) {
    int kk = kk0 + i;
    float v = 0.f;
    if (kk < Cin) {
      v = xp[(size_t)kk * chs];
      if (relu) v = fmaxf(v, 0.f);
    }
    o[i] = (_Float16)v;
  }
  *(v4h*)(Xp + (size_t)m * Kp + kk0) = o;
}

// =====================================================================
// pack_w: f32 weights W[Cout][Cin] -> f16 A fragments in lane layout.
//   frag t = ct*Kt + kt; element layout for A 16x32, this code's contiguous k order:
//   lane<16: M=lane, K = {0..15}; lane>=16: M=lane-16, K={16..31}
// One thread per (frag, lane) writes 16 f16 (32B).
// =====================================================================
__global__ void pack_w_kernel(
    const float* __restrict__ W, _Float16* __restrict__ Ap,
    int Cout, int Cin, int Ct, int Kt)
{
  int t = blockIdx.x * blockDim.x + threadIdx.x;
  int total = Ct * Kt * 32;
  if (t >= total) return;
  int lane = t & 31;
  int frag = t >> 5;
  int kt = frag % Kt;
  int ct = frag / Kt;
  int half = lane >> 4;
  int row  = ct * 16 + (lane & 15);

  v16h o;
#pragma unroll
  for (int i = 0; i < 16; ++i) {
    int k  = half * 16 + i;
    int ci = kt * 32 + k;
    float v = (row < Cout && ci < Cin) ? W[(size_t)row * Cin + ci] : 0.f;
    o[i] = (_Float16)v;
  }
  *(v16h*)(Ap + ((size_t)t << 4)) = o;
}

// =====================================================================
// WMMA GEMM on packed operands, specialized on the number of Cout tiles
// (CT compile-time -> no branches around WMMA, accumulators stay pinned):
//   Y[b, cbase+co, sp] = sum_kk Ap[co,kk] * Xp[m,kk]
// One wave = one 16-pixel M tile; B frag reused across all CT tiles.
// B frag = one 32B contiguous load; A frag = one 32B contiguous load.
// =====================================================================
template <int CT>
__global__ __launch_bounds__(256) void pw_gemm_wmma(
    const _Float16* __restrict__ Xp, const _Float16* __restrict__ Ap,
    float* __restrict__ Y,
    int Bn, int Kp, int Cout, int cbase, int Cy, int HWo)
{
  const int lane = threadIdx.x & 31;
  const int wave = threadIdx.x >> 5;
  const int Mt   = (Bn * HWo) >> 4;
  const int mt   = blockIdx.x * 8 + wave;
  if (mt >= Mt) return;                    // wave-uniform exit

  const int m0   = mt << 4;
  const int col  = lane & 15;
  const int half = lane >> 4;
  const int Kt   = Kp >> 5;

  const _Float16* brow  = Xp + (size_t)(m0 + col) * Kp + half * 16;
  const _Float16* afrag = Ap + ((size_t)lane << 4);

  v8f acc[CT];
#pragma unroll
  for (int ct = 0; ct < CT; ++ct) {
    v8f z = {0.f, 0.f, 0.f, 0.f, 0.f, 0.f, 0.f, 0.f};
    acc[ct] = z;
  }

  for (int kt = 0; kt < Kt; ++kt) {
    v16h bm = *(const v16h*)(brow + (kt << 5));
#pragma unroll
    for (int ct = 0; ct < CT; ++ct) {
      v16h am = *(const v16h*)(afrag + ((size_t)(ct * Kt + kt) << 9));
      acc[ct] = __builtin_amdgcn_wmma_f32_16x16x32_f16(
          false, am, false, bm, (short)0, acc[ct], false, false);
    }
  }

  const int b  = m0 / HWo;
  const int sp = (m0 - b * HWo) + col;
  float* ycol = Y + ((size_t)b * Cy + cbase) * HWo + sp;
#pragma unroll
  for (int ct = 0; ct < CT; ++ct) {
#pragma unroll
    for (int r = 0; r < 8; ++r) {
      int row = ct * 16 + (half ? 8 + r : r);
      if (row < Cout) ycol[(size_t)row * HWo] = acc[ct][r];
    }
  }
}

// =====================================================================
// Depthwise conv (groups == C), optional ReLU on input, dilation support
// =====================================================================
__global__ __launch_bounds__(256) void dwconv_kernel(
    const float* __restrict__ X, const float* __restrict__ W, float* __restrict__ Y,
    int Bn, int C, int Hin, int Win, int Hout, int Wout,
    int k, int stride, int pad, int dil, int relu_in)
{
  int idx = blockIdx.x * blockDim.x + threadIdx.x;
  int total = Bn * C * Hout * Wout;
  if (idx >= total) return;
  int wo = idx % Wout; int t = idx / Wout;
  int ho = t % Hout;   t /= Hout;
  int c  = t % C;      int b = t / C;

  const float* xp = X + ((size_t)b * C + c) * Hin * Win;
  const float* wp = W + (size_t)c * k * k;
  float acc = 0.f;
  for (int r = 0; r < k; ++r) {
    int hi = ho * stride - pad + r * dil;
    if (hi < 0 || hi >= Hin) continue;
    for (int s = 0; s < k; ++s) {
      int wi = wo * stride - pad + s * dil;
      if (wi < 0 || wi >= Win) continue;
      float v = xp[(size_t)hi * Win + wi];
      if (relu_in) v = fmaxf(v, 0.f);
      acc += v * wp[r * k + s];
    }
  }
  Y[idx] = acc;
}

// =====================================================================
// 3x3 pool, pad 1; is_max=1 -> max pool, else avg (count_include_pad=False)
// =====================================================================
__global__ __launch_bounds__(256) void pool3_kernel(
    const float* __restrict__ X, float* __restrict__ Y,
    int Bn, int C, int Hin, int Win, int Hout, int Wout, int stride, int is_max)
{
  int idx = blockIdx.x * blockDim.x + threadIdx.x;
  int total = Bn * C * Hout * Wout;
  if (idx >= total) return;
  int wo = idx % Wout; int t = idx / Wout;
  int ho = t % Hout;   t /= Hout;
  int c  = t % C;      int b = t / C;

  const float* xp = X + ((size_t)b * C + c) * Hin * Win;
  float best = -3.4e38f, sum = 0.f;
  int cnt = 0;
  for (int r = 0; r < 3; ++r) {
    int hi = ho * stride - 1 + r;
    if (hi < 0 || hi >= Hin) continue;
    for (int s = 0; s < 3; ++s) {
      int wi = wo * stride - 1 + s;
      if (wi < 0 || wi >= Win) continue;
      float v = xp[(size_t)hi * Win + wi];
      best = fmaxf(best, v);
      sum += v; ++cnt;
    }
  }
  Y[idx] = is_max ? best : sum / (float)cnt;
}

// =====================================================================
// BN batch statistics: one block per channel, sum & sumsq over (B,H,W)
// =====================================================================
__global__ __launch_bounds__(256) void bn_stats_kernel(
    const float* __restrict__ X, float* __restrict__ stats, int Bn, int C, int HW)
{
  __shared__ float ss[256], sq[256];
  int c = blockIdx.x;
  float s = 0.f, q = 0.f;
  int M = Bn * HW;
  for (int i = threadIdx.x; i < M; i += 256) {
    int b = i / HW, sp = i - b * HW;
    float v = X[((size_t)b * C + c) * HW + sp];
    s += v; q += v * v;
  }
  ss[threadIdx.x] = s; sq[threadIdx.x] = q;
  __syncthreads();
  for (int st = 128; st > 0; st >>= 1) {
    if (threadIdx.x < st) { ss[threadIdx.x] += ss[threadIdx.x + st]; sq[threadIdx.x] += sq[threadIdx.x + st]; }
    __syncthreads();
  }
  if (threadIdx.x == 0) { stats[2 * c] = ss[0]; stats[2 * c + 1] = sq[0]; }
}

// =====================================================================
// BN apply; optional affine and optional fused alpha-accumulate.
// =====================================================================
__global__ __launch_bounds__(256) void bn_apply_kernel(
    const float* __restrict__ X, float* __restrict__ Out, const float* __restrict__ stats,
    const float* __restrict__ gamma, const float* __restrict__ beta,
    const float* __restrict__ alpha, int ai, int accumulate,
    int Bn, int C, int HW)
{
  int idx = blockIdx.x * blockDim.x + threadIdx.x;
  int total = Bn * C * HW;
  if (idx >= total) return;
  int c = (idx / HW) % C;
  float invM = 1.f / (float)(Bn * HW);
  float mean = stats[2 * c] * invM;
  float var  = stats[2 * c + 1] * invM - mean * mean;
  float y = (X[idx] - mean) * rsqrtf(var + 1e-5f);
  if (gamma) y = y * gamma[c] + beta[c];
  float a = alpha ? alpha[ai] : 1.f;
  if (accumulate) Out[idx] += a * y;
  else            Out[idx]  = a * y;
}

__global__ __launch_bounds__(256) void axpy_kernel(
    const float* __restrict__ X, float* __restrict__ Out,
    const float* __restrict__ alpha, int ai, int n)
{
  int i = blockIdx.x * blockDim.x + threadIdx.x;
  if (i < n) Out[i] += alpha[ai] * X[i];
}

__global__ __launch_bounds__(256) void fill0_kernel(float* __restrict__ p, int n)
{
  int i = blockIdx.x * blockDim.x + threadIdx.x;
  if (i < n) p[i] = 0.f;
}

// Copy (B,Csub,HW) tensor into channel slice [cbase, cbase+Csub) of (B,Ctot,HW)
__global__ __launch_bounds__(256) void concat_kernel(
    const float* __restrict__ src, float* __restrict__ dst,
    int Bn, int Csub, int HW, int Ctot, int cbase)
{
  int idx = blockIdx.x * blockDim.x + threadIdx.x;
  int total = Bn * Csub * HW;
  if (idx >= total) return;
  int sp = idx % HW; int t = idx / HW;
  int c = t % Csub;  int b = t / Csub;
  dst[((size_t)b * Ctot + cbase + c) * HW + sp] = src[idx];
}

// Stem direct 3x3 conv, pad 1, stride 1 (Cin=3, cheap)
__global__ __launch_bounds__(256) void conv3x3_kernel(
    const float* __restrict__ X, const float* __restrict__ W, float* __restrict__ Y,
    int Bn, int Cin, int Cout, int H, int Wd)
{
  int idx = blockIdx.x * blockDim.x + threadIdx.x;
  int total = Bn * Cout * H * Wd;
  if (idx >= total) return;
  int w = idx % Wd; int t = idx / Wd;
  int h = t % H;    t /= H;
  int co = t % Cout; int b = t / Cout;
  float acc = 0.f;
  for (int ci = 0; ci < Cin; ++ci) {
    const float* xp = X + ((size_t)b * Cin + ci) * H * Wd;
    const float* wp = W + ((size_t)co * Cin + ci) * 9;
    for (int r = 0; r < 3; ++r) {
      int hi = h - 1 + r;
      if (hi < 0 || hi >= H) continue;
      for (int s = 0; s < 3; ++s) {
        int wi = w - 1 + s;
        if (wi < 0 || wi >= Wd) continue;
        acc += xp[(size_t)hi * Wd + wi] * wp[r * 3 + s];
      }
    }
  }
  Y[idx] = acc;
}

// Row softmax over 8 columns
__global__ void softmax8_kernel(const float* __restrict__ a, float* __restrict__ o, int rows)
{
  int r = blockIdx.x * blockDim.x + threadIdx.x;
  if (r >= rows) return;
  float m = a[r * 8];
  for (int i = 1; i < 8; ++i) m = fmaxf(m, a[r * 8 + i]);
  float e[8], s = 0.f;
  for (int i = 0; i < 8; ++i) { e[i] = __expf(a[r * 8 + i] - m); s += e[i]; }
  for (int i = 0; i < 8; ++i) o[r * 8 + i] = e[i] / s;
}

// Global average pool + FC. One block per batch element. C <= 256.
__global__ __launch_bounds__(256) void gap_fc_kernel(
    const float* __restrict__ X, const float* __restrict__ fw,
    const float* __restrict__ fb, float* __restrict__ out,
    int Bn, int C, int HW)
{
  __shared__ float feat[256];
  int b = blockIdx.x;
  for (int c = threadIdx.x; c < C; c += 256) {
    const float* xp = X + ((size_t)b * C + c) * HW;
    float s = 0.f;
    for (int i = 0; i < HW; ++i) s += xp[i];
    feat[c] = s / (float)HW;
  }
  __syncthreads();
  for (int cls = threadIdx.x; cls < 10; cls += 256) {
    float s = fb[cls];
    for (int c = 0; c < C; ++c) s += feat[c] * fw[(size_t)cls * C + c];
    out[b * 10 + cls] = s;
  }
}

// =====================================================================
// Host orchestration
// =====================================================================
namespace {

constexpr int BATCH = 32;

struct EdgeP {
  const float *dil3_dw, *dil3_pw, *dil5_dw, *dil5_pw;
  const float *sep3_dw1, *sep3_dw2, *sep3_pw1, *sep3_pw2;
  const float *sep5_dw1, *sep5_dw2, *sep5_pw1, *sep5_pw2;
  const float *skip_fr1, *skip_fr2;
};
struct CellP {
  const float *pre0_conv, *pre0_fr1, *pre0_fr2, *pre1_conv;
  EdgeP edges[14];
};

struct Scratch {
  _Float16* Xp;
  _Float16* Ap;
  float* stats;
};

static void pw(const float* X, const float* W, float* Y,
               int Cin, int Hin, int Cout, int cbase, int Cy, int Hout,
               int stride, int off, int relu, const Scratch& sc, hipStream_t s)
{
  const int Kp = cdiv(Cin, 32) * 32;
  const int Kt = Kp / 32;
  const int Ct = cdiv(Cout, 16);
  const int M  = BATCH * Hout * Hout;
  const int tot4 = M * (Kp / 4);
  pack_x_kernel<<<cdiv(tot4, TPB), TPB, 0, s>>>(X, sc.Xp, BATCH, Cin, Hin, Hin,
                                                Hout, Hout, stride, off, relu, Kp);
  pack_w_kernel<<<cdiv(Ct * Kt * 32, 64), 64, 0, s>>>(W, sc.Ap, Cout, Cin, Ct, Kt);
  const int HWo = Hout * Hout;
  const int grid = cdiv(M / 16, 8);
  switch (Ct) {
    case 1: pw_gemm_wmma<1><<<grid, TPB, 0, s>>>(sc.Xp, sc.Ap, Y, BATCH, Kp, Cout, cbase, Cy, HWo); break;
    case 2: pw_gemm_wmma<2><<<grid, TPB, 0, s>>>(sc.Xp, sc.Ap, Y, BATCH, Kp, Cout, cbase, Cy, HWo); break;
    case 3: pw_gemm_wmma<3><<<grid, TPB, 0, s>>>(sc.Xp, sc.Ap, Y, BATCH, Kp, Cout, cbase, Cy, HWo); break;
    default: pw_gemm_wmma<4><<<grid, TPB, 0, s>>>(sc.Xp, sc.Ap, Y, BATCH, Kp, Cout, cbase, Cy, HWo); break;
  }
}

static void bn(const float* X, float* Out, float* stats, const float* g, const float* bb,
               const float* alpha, int ai, int accum, int C, int HW, hipStream_t s)
{
  bn_stats_kernel<<<C, TPB, 0, s>>>(X, stats, BATCH, C, HW);
  int total = BATCH * C * HW;
  bn_apply_kernel<<<cdiv(total, TPB), TPB, 0, s>>>(X, Out, stats, g, bb, alpha, ai, accum, BATCH, C, HW);
}

static void mixed_op(const float* x, float* acc, const EdgeP& ep,
                     int C, int Hs, int Ho, int stride,
                     const float* wsm, int edge,
                     float* tmp0, float* tmp1, const Scratch& sc, hipStream_t s)
{
  const int HWo = Ho * Ho;
  const int nOut = BATCH * C * HWo;
  const int base = edge * 8;
  float* stats = sc.stats;

  // op1: max_pool -> bn -> +=
  pool3_kernel<<<cdiv(nOut, TPB), TPB, 0, s>>>(x, tmp0, BATCH, C, Hs, Hs, Ho, Ho, stride, 1);
  bn(tmp0, acc, stats, nullptr, nullptr, wsm, base + 1, 1, C, HWo, s);
  // op2: avg_pool -> bn -> +=
  pool3_kernel<<<cdiv(nOut, TPB), TPB, 0, s>>>(x, tmp0, BATCH, C, Hs, Hs, Ho, Ho, stride, 0);
  bn(tmp0, acc, stats, nullptr, nullptr, wsm, base + 2, 1, C, HWo, s);
  // op3: skip / factorized reduce
  if (stride == 1) {
    axpy_kernel<<<cdiv(nOut, TPB), TPB, 0, s>>>(x, acc, wsm, base + 3, nOut);
  } else {
    pw(x, ep.skip_fr1, tmp0, C, Hs, C / 2, 0,     C, Ho, 2, 0, 1, sc, s);
    pw(x, ep.skip_fr2, tmp0, C, Hs, C / 2, C / 2, C, Ho, 2, 1, 1, sc, s);
    bn(tmp0, acc, stats, nullptr, nullptr, wsm, base + 3, 1, C, HWo, s);
  }
  // op4/op5: sep_conv k=3,5
  for (int t = 0; t < 2; ++t) {
    int k = t ? 5 : 3, pad = (k - 1) / 2;
    const float* dw1 = t ? ep.sep5_dw1 : ep.sep3_dw1;
    const float* dw2 = t ? ep.sep5_dw2 : ep.sep3_dw2;
    const float* pw1 = t ? ep.sep5_pw1 : ep.sep3_pw1;
    const float* pw2 = t ? ep.sep5_pw2 : ep.sep3_pw2;
    dwconv_kernel<<<cdiv(nOut, TPB), TPB, 0, s>>>(x, dw1, tmp0, BATCH, C, Hs, Hs, Ho, Ho, k, stride, pad, 1, 1);
    pw(tmp0, pw1, tmp1, C, Ho, C, 0, C, Ho, 1, 0, 0, sc, s);
    bn(tmp1, tmp1, stats, nullptr, nullptr, nullptr, 0, 0, C, HWo, s);       // in-place bn
    dwconv_kernel<<<cdiv(nOut, TPB), TPB, 0, s>>>(tmp1, dw2, tmp0, BATCH, C, Ho, Ho, Ho, Ho, k, 1, pad, 1, 1);
    pw(tmp0, pw2, tmp1, C, Ho, C, 0, C, Ho, 1, 0, 0, sc, s);
    bn(tmp1, acc, stats, nullptr, nullptr, wsm, base + 4 + t, 1, C, HWo, s);
  }
  // op6/op7: dil_conv k=3,5 (dilation 2, pad k-1)
  for (int t = 0; t < 2; ++t) {
    int k = t ? 5 : 3, pad = k - 1;
    const float* dwW = t ? ep.dil5_dw : ep.dil3_dw;
    const float* pwW = t ? ep.dil5_pw : ep.dil3_pw;
    dwconv_kernel<<<cdiv(nOut, TPB), TPB, 0, s>>>(x, dwW, tmp0, BATCH, C, Hs, Hs, Ho, Ho, k, stride, pad, 2, 1);
    pw(tmp0, pwW, tmp1, C, Ho, C, 0, C, Ho, 1, 0, 0, sc, s);
    bn(tmp1, acc, stats, nullptr, nullptr, wsm, base + 6 + t, 1, C, HWo, s);
  }
}

} // namespace

extern "C" void kernel_launch(void* const* d_in, const int* in_sizes, int n_in,
                              void* d_out, int out_size, void* d_ws, size_t ws_size,
                              hipStream_t stream)
{
  (void)in_sizes; (void)n_in; (void)out_size; (void)ws_size;

  // ---------- walk flattened input pytree ----------
  // top-level insertion order: x, params, normal_alphas, reduce_alphas
  // params: sorted keys -> cells, fc_b, fc_w, stem_b, stem_g, stem_w
  // cell dict: ops, pre0, pre1 ; edge dict: dil3, dil5, sep3, sep5, [skip]
  int wi = 0;
  const float* x = (const float*)d_in[wi++];
  CellP cells[8];
  {
    bool red_prev = false;
    for (int c = 0; c < 8; ++c) {
      bool red = (c == 2 || c == 5);
      int e = 0;
      for (int i = 0; i < 4; ++i) {
        for (int j = 0; j < i + 2; ++j) {
          bool s2 = red && j < 2;
          EdgeP& ep = cells[c].edges[e++];
          ep.dil3_dw  = (const float*)d_in[wi++]; ep.dil3_pw  = (const float*)d_in[wi++];
          ep.dil5_dw  = (const float*)d_in[wi++]; ep.dil5_pw  = (const float*)d_in[wi++];
          ep.sep3_dw1 = (const float*)d_in[wi++]; ep.sep3_dw2 = (const float*)d_in[wi++];
          ep.sep3_pw1 = (const float*)d_in[wi++]; ep.sep3_pw2 = (const float*)d_in[wi++];
          ep.sep5_dw1 = (const float*)d_in[wi++]; ep.sep5_dw2 = (const float*)d_in[wi++];
          ep.sep5_pw1 = (const float*)d_in[wi++]; ep.sep5_pw2 = (const float*)d_in[wi++];
          if (s2) { ep.skip_fr1 = (const float*)d_in[wi++]; ep.skip_fr2 = (const float*)d_in[wi++]; }
          else    { ep.skip_fr1 = ep.skip_fr2 = nullptr; }
        }
      }
      if (red_prev) {
        cells[c].pre0_conv = nullptr;
        cells[c].pre0_fr1 = (const float*)d_in[wi++];
        cells[c].pre0_fr2 = (const float*)d_in[wi++];
      } else {
        cells[c].pre0_conv = (const float*)d_in[wi++];
        cells[c].pre0_fr1 = cells[c].pre0_fr2 = nullptr;
      }
      cells[c].pre1_conv = (const float*)d_in[wi++];
      red_prev = red;
    }
  }
  const float* fc_b   = (const float*)d_in[wi++];
  const float* fc_w   = (const float*)d_in[wi++];
  const float* stem_b = (const float*)d_in[wi++];
  const float* stem_g = (const float*)d_in[wi++];
  const float* stem_w = (const float*)d_in[wi++];
  const float* normal_a = (const float*)d_in[wi++];
  const float* reduce_a = (const float*)d_in[wi++];

  // ---------- workspace carve-up (float units; all offsets 32B aligned) ----------
  const size_t SLOT = (size_t)32 * 64 * 32 * 32;   // 2.10M floats (max state)
  const size_t PREB = (size_t)32 * 32 * 32 * 32;   // 1.05M floats
  const size_t NODE = (size_t)32 * 16 * 32 * 32;   // 0.52M floats
  const size_t XPEL = (size_t)32 * 1024 * 64;      // 2.10M f16 = 1.05M floats (max M*Kp)
  float* p = (float*)d_ws;
  float* slots[3]; for (int i = 0; i < 3; ++i) { slots[i] = p; p += SLOT; }
  float* pre0b = p; p += PREB;
  float* pre1b = p; p += PREB;
  float* node[4]; for (int i = 0; i < 4; ++i) { node[i] = p; p += NODE; }
  float* tmp0 = p; p += NODE;
  float* tmp1 = p; p += NODE;
  Scratch sc;
  sc.stats = p; p += 512;
  float* wnorm = p; p += 128;
  float* wred  = p; p += 128;
  sc.Xp = (_Float16*)p; p += XPEL / 2;
  sc.Ap = (_Float16*)p; p += 16384;   // 32K f16 fragments

  // ---------- alpha softmax ----------
  softmax8_kernel<<<1, 32, 0, stream>>>(normal_a, wnorm, 14);
  softmax8_kernel<<<1, 32, 0, stream>>>(reduce_a, wred, 14);

  // ---------- stem: conv3x3 (3->48) + affine BN ----------
  {
    int total = BATCH * 48 * 32 * 32;
    conv3x3_kernel<<<cdiv(total, TPB), TPB, 0, stream>>>(x, stem_w, slots[0], BATCH, 3, 48, 32, 32);
    bn(slots[0], slots[0], sc.stats, stem_g, stem_b, nullptr, 0, 0, 48, 1024, stream);
  }

  // ---------- cells ----------
  const float* s0 = slots[0];
  const float* s1 = slots[0];
  int outi = 1;
  int C_pp = 48, C_p = 48, C_cur = 16;
  int H_pp = 32, H_p = 32;
  bool red_prev = false;

  for (int ci = 0; ci < 8; ++ci) {
    const bool red = (ci == 2 || ci == 5);
    if (red) C_cur *= 2;
    const int Hc = H_p;
    const int Ho = red ? Hc / 2 : Hc;
    const int HWc = Hc * Hc, HWo = Ho * Ho;
    float* outb = slots[outi];
    const float* wsm = red ? wred : wnorm;
    const CellP& cp = cells[ci];

    // preprocess s0
    if (red_prev) {
      pw(s0, cp.pre0_fr1, pre0b, C_pp, H_pp, C_cur / 2, 0,         C_cur, Hc, 2, 0, 1, sc, stream);
      pw(s0, cp.pre0_fr2, pre0b, C_pp, H_pp, C_cur / 2, C_cur / 2, C_cur, Hc, 2, 1, 1, sc, stream);
    } else {
      pw(s0, cp.pre0_conv, pre0b, C_pp, H_pp, C_cur, 0, C_cur, Hc, 1, 0, 1, sc, stream);
    }
    bn(pre0b, pre0b, sc.stats, nullptr, nullptr, nullptr, 0, 0, C_cur, HWc, stream);
    // preprocess s1
    pw(s1, cp.pre1_conv, pre1b, C_p, Hc, C_cur, 0, C_cur, Hc, 1, 0, 1, sc, stream);
    bn(pre1b, pre1b, sc.stats, nullptr, nullptr, nullptr, 0, 0, C_cur, HWc, stream);

    // nodes
    const float* statesp[6] = { pre0b, pre1b, node[0], node[1], node[2], node[3] };
    int e = 0;
    for (int i = 0; i < 4; ++i) {
      int nAcc = BATCH * C_cur * HWo;
      fill0_kernel<<<cdiv(nAcc, TPB), TPB, 0, stream>>>(node[i], nAcc);
      for (int j = 0; j < i + 2; ++j) {
        int stride = (red && j < 2) ? 2 : 1;
        int Hs = (j < 2) ? Hc : Ho;
        mixed_op(statesp[j], node[i], cp.edges[e], C_cur, Hs, Ho, stride,
                 wsm, e, tmp0, tmp1, sc, stream);
        ++e;
      }
    }

    // concat nodes -> cell output
    for (int i = 0; i < 4; ++i) {
      int total = BATCH * C_cur * HWo;
      concat_kernel<<<cdiv(total, TPB), TPB, 0, stream>>>(node[i], outb, BATCH, C_cur, HWo, 4 * C_cur, i * C_cur);
    }

    // rotate state
    s0 = s1; s1 = outb; outi = (outi + 1) % 3;
    C_pp = C_p; C_p = 4 * C_cur;
    H_pp = Hc;  H_p = Ho;
    red_prev = red;
  }

  // ---------- head: GAP + FC ----------
  gap_fc_kernel<<<BATCH, TPB, 0, stream>>>(s1, fc_w, fc_b, (float*)d_out, BATCH, C_p, H_p * H_p);
}
